// MapFormerEM_35141422416175
// MI455X (gfx1250) — matrix operand order repaired and checked by reading, not compiled, not run on hardware
//
#include <hip/hip_runtime.h>
#include <math.h>

typedef unsigned short u16;

// ---------------- types for WMMA ----------------
typedef __bf16  v16bf __attribute__((ext_vector_type(16)));
typedef float   v8f   __attribute__((ext_vector_type(8)));

union FragU { uint4 u[2]; v16bf v; };

static __device__ __forceinline__ u16 f2bf(float f) {
  union { float f; unsigned u; } c; c.f = f;
  unsigned u = c.u;
  unsigned r = (u + 0x7FFFu + ((u >> 16) & 1u)) >> 16;   // RNE
  return (u16)r;
}

// A-operand fragment (16xK32 bf16): lanes0-15 K{0..7,16..23}, lanes16-31 K{8..15,24..31}
static __device__ __forceinline__ v16bf load_frag_a(const u16* base, int stride,
                                                    int l15, int half) {
  const u16* row = base + l15 * stride;
  FragU f;
  f.u[0] = *reinterpret_cast<const uint4*>(row + (half ? 8 : 0));
  f.u[1] = *reinterpret_cast<const uint4*>(row + (half ? 24 : 16));
  return f.v;
}
// B-operand fragment (K32x16 bf16), tile stored [n][k]: lanes0-15 K{0..7,16..23}, lanes16-31 K{8..15,24..31}
static __device__ __forceinline__ v16bf load_frag_b(const u16* base, int stride,
                                                    int l15, int half) {
  const u16* row = base + l15 * stride;
  FragU f;
  f.u[0] = *reinterpret_cast<const uint4*>(row + half * 8);
  f.u[1] = *reinterpret_cast<const uint4*>(row + half * 8 + 16);
  return f.v;
}

static __device__ __forceinline__ v8f wmma_bf16(v16bf a, v16bf b, v8f c) {
  return __builtin_amdgcn_wmma_f32_16x16x32_bf16(false, a, false, b, (short)0, c, false, false);
}

// ---------------- constants ----------------
#define BB 2
#define TT 2048
#define DD 1024
#define HH 16
#define DH 64
#define RR 32
#define MM 4096          // B*T
#define OV 8192

// ---------------- small kernels ----------------

// x[row] = obs_emb[obs[row]]  (f32 residual stream)
__global__ __launch_bounds__(256) void k_embed(const int* __restrict__ obs,
                                               const float* __restrict__ emb,
                                               float* __restrict__ x) {
  int row = blockIdx.x;
  int o = obs[row];
  const float* src = emb + (size_t)o * DD;
  float* dst = x + (size_t)row * DD;
  for (int c = threadIdx.x; c < DD; c += 256) dst[c] = src[c];
}

// delta[row][r] = dot(action_emb[a], lie_W[:,r]) + lie_b[r]
__global__ __launch_bounds__(256) void k_delta(const int* __restrict__ actions,
                                               const float* __restrict__ aemb,
                                               const float* __restrict__ lieW,
                                               const float* __restrict__ lieb,
                                               float* __restrict__ delta) {
  int row = blockIdx.x;
  int a = actions[row];
  int r = threadIdx.x & 31;
  int seg = threadIdx.x >> 5;
  const float* e = aemb + (size_t)a * DD + seg * 128;
  const float* w = lieW + (size_t)(seg * 128) * RR + r;
  float p = 0.f;
  #pragma unroll 4
  for (int j = 0; j < 128; ++j) p += e[j] * w[j * RR];
  __shared__ float red[8][32];
  red[seg][r] = p;
  __syncthreads();
  if (threadIdx.x < 32) {
    float s = lieb[r];
    #pragma unroll
    for (int q = 0; q < 8; ++q) s += red[q][r];
    delta[(size_t)row * RR + r] = s;
  }
}

// cumulative angle -> cs[row] = [cos th_0..31, sin th_0..31] (bf16)
__global__ void k_scan(const float* __restrict__ delta, u16* __restrict__ cs) {
  int t = threadIdx.x;            // 64 threads
  int b = t >> 5, r = t & 31;
  float th = 0.f;
  for (int i = 0; i < TT; ++i) {
    size_t row = (size_t)b * TT + i;
    th += delta[row * RR + r];
    float s, c;
    sincosf(th, &s, &c);
    cs[row * 64 + r] = f2bf(c);
    cs[row * 64 + 32 + r] = f2bf(s);
  }
}

// W_eff transposed [n][r] (bf16): rows 0..31 cos-combination, 32..63 sin-combination
__global__ __launch_bounds__(256) void k_weff(const float* __restrict__ posW,
                                              u16* __restrict__ wefft) {
  int r = blockIdx.x;                 // 0..63
  int i = r & 31;
  size_t d0 = (size_t)((2 * i) * 64 + 2 * i) * DD;
  size_t d1 = (size_t)((2 * i + 1) * 64 + (2 * i + 1)) * DD;
  size_t o0 = (size_t)((2 * i + 1) * 64 + 2 * i) * DD;
  size_t o1 = (size_t)((2 * i) * 64 + (2 * i + 1)) * DD;
  for (int c = threadIdx.x; c < DD; c += 256) {
    float v = (r < 32) ? (posW[d0 + c] + posW[d1 + c])
                       : (posW[o0 + c] - posW[o1 + c]);
    wefft[(size_t)c * 64 + r] = f2bf(v);
  }
}

// weight transpose + f32->bf16: Wt[n][k] = W[k][n]
__global__ __launch_bounds__(256) void k_wtrans(const float* __restrict__ W,
                                                u16* __restrict__ Wt, int N, int K) {
  __shared__ float tile[32][33];
  int n0 = blockIdx.x * 32, k0 = blockIdx.y * 32;
  int tx = threadIdx.x & 31, ty = threadIdx.x >> 5;   // 8 rows per pass
  #pragma unroll
  for (int i = 0; i < 4; ++i)
    tile[ty + 8 * i][tx] = W[(size_t)(k0 + ty + 8 * i) * N + n0 + tx];
  __syncthreads();
  #pragma unroll
  for (int i = 0; i < 4; ++i)
    Wt[(size_t)(n0 + ty + 8 * i) * K + k0 + tx] = f2bf(tile[tx][ty + 8 * i]);
}

// LayerNorm over last dim (1024); f32 in, bf16 out
__global__ __launch_bounds__(256) void k_ln(const float* __restrict__ x,
                                            const float* __restrict__ g,
                                            const float* __restrict__ b,
                                            u16* __restrict__ out) {
  int row = blockIdx.x;
  const float* xr = x + (size_t)row * DD;
  float v[4], s1 = 0.f, s2 = 0.f;
  #pragma unroll
  for (int i = 0; i < 4; ++i) {
    v[i] = xr[threadIdx.x + 256 * i];
    s1 += v[i]; s2 += v[i] * v[i];
  }
  for (int m = 16; m > 0; m >>= 1) {
    s1 += __shfl_xor(s1, m, 32);
    s2 += __shfl_xor(s2, m, 32);
  }
  __shared__ float sh[2][8];
  int wid = threadIdx.x >> 5;
  if ((threadIdx.x & 31) == 0) { sh[0][wid] = s1; sh[1][wid] = s2; }
  __syncthreads();
  s1 = 0.f; s2 = 0.f;
  #pragma unroll
  for (int q = 0; q < 8; ++q) { s1 += sh[0][q]; s2 += sh[1][q]; }
  float mu = s1 * (1.f / DD);
  float var = s2 * (1.f / DD) - mu * mu;
  float rs = rsqrtf(var + 1e-5f);
  #pragma unroll
  for (int i = 0; i < 4; ++i) {
    int c = threadIdx.x + 256 * i;
    out[(size_t)row * DD + c] = f2bf((v[i] - mu) * rs * g[c] + b[c]);
  }
}

// ---------------- tiled WMMA GEMM (bf16 in, f32 accum) ----------------
// A: bf16 [M][lda]; Wt: bf16 pre-transposed [N][ldb=K]. Block tile 128x128, K-step 32,
// 8 waves (4Mx2N), software-pipelined global->regs->LDS copies.
// mode: 0 plain, 1 exact GELU, 2 += residual(f32). obf: bf16 output. hsplit: head interleave.
__global__ __launch_bounds__(256) void k_gemm(const u16* __restrict__ A, int lda,
                                              const u16* __restrict__ Wt, int ldb,
                                              const float* __restrict__ bias,
                                              const float* __restrict__ residual,
                                              void* __restrict__ Cv, int ldc,
                                              int K, int mode, int hsplit, int qoff,
                                              float scale, int obf) {
  __shared__ __align__(16) u16 As[128][40];
  __shared__ __align__(16) u16 Bs[128][40];

  const int tid = threadIdx.x;
  const int lane = tid & 31, wid = tid >> 5;
  const int l15 = lane & 15, half = lane >> 4;
  const int wm = wid & 3, wn = wid >> 2;
  const int m0 = blockIdx.y * 128, n0 = blockIdx.x * 128;

  // copy-thread mapping: 4 chunks of 8 bf16 per 32-k row
  const int cc = tid & 3, rr = tid >> 2;          // rr in 0..63
  const u16* Ap0 = A + (size_t)(m0 + rr) * lda + cc * 8;
  const u16* Ap1 = Ap0 + (size_t)64 * lda;
  const u16* Bp0 = Wt + (size_t)(n0 + rr) * ldb + cc * 8;
  const u16* Bp1 = Bp0 + (size_t)64 * ldb;

  v8f acc[2][4];
  #pragma unroll
  for (int mi = 0; mi < 2; ++mi)
    #pragma unroll
    for (int ni = 0; ni < 4; ++ni)
      #pragma unroll
      for (int j = 0; j < 8; ++j) acc[mi][ni][j] = 0.f;

  uint4 ra0 = *(const uint4*)(Ap0);
  uint4 ra1 = *(const uint4*)(Ap1);
  uint4 rb0 = *(const uint4*)(Bp0);
  uint4 rb1 = *(const uint4*)(Bp1);

  for (int k0 = 0; k0 < K; k0 += 32) {
    __syncthreads();
    *(uint4*)&As[rr][cc * 8] = ra0;
    *(uint4*)&As[rr + 64][cc * 8] = ra1;
    *(uint4*)&Bs[rr][cc * 8] = rb0;
    *(uint4*)&Bs[rr + 64][cc * 8] = rb1;
    __syncthreads();
    if (k0 + 32 < K) {   // next tile's loads overlap the WMMA block below
      ra0 = *(const uint4*)(Ap0 + k0 + 32);
      ra1 = *(const uint4*)(Ap1 + k0 + 32);
      rb0 = *(const uint4*)(Bp0 + k0 + 32);
      rb1 = *(const uint4*)(Bp1 + k0 + 32);
    }

    v16bf af[2], bfv[4];
    af[0] = load_frag_a(&As[wm * 32][0], 40, l15, half);
    af[1] = load_frag_a(&As[wm * 32 + 16][0], 40, l15, half);
    #pragma unroll
    for (int ni = 0; ni < 4; ++ni)
      bfv[ni] = load_frag_b(&Bs[wn * 64 + ni * 16][0], 40, l15, half);
    #pragma unroll
    for (int mi = 0; mi < 2; ++mi)
      #pragma unroll
      for (int ni = 0; ni < 4; ++ni)
        acc[mi][ni] = wmma_bf16(af[mi], bfv[ni], acc[mi][ni]);
  }

  // epilogue
  #pragma unroll
  for (int mi = 0; mi < 2; ++mi) {
    #pragma unroll
    for (int ni = 0; ni < 4; ++ni) {
      #pragma unroll
      for (int j = 0; j < 8; ++j) {
        int row = m0 + wm * 32 + mi * 16 + j + half * 8;
        int c = n0 + wn * 64 + ni * 16 + l15;
        float v = acc[mi][ni][j];
        if (bias) v += bias[c];
        v *= scale;
        if (mode == 1) v = 0.5f * v * (1.0f + erff(v * 0.70710678118654752f));
        int oc = hsplit ? (((c >> 6) << 7) + (c & 63) + qoff) : c;
        size_t oidx = (size_t)row * ldc + oc;
        if (mode == 2) v += residual[oidx];
        if (obf) ((u16*)Cv)[oidx] = f2bf(v);
        else     ((float*)Cv)[oidx] = v;
      }
    }
  }
}

// ---------------- flash attention (causal) ----------------
// Q,K: bf16 (B*T, H*128) (Q pre-scaled by 1/8 in projection), V: bf16 (B*T, H*64).
// Block = (qblock 128, head, batch); 8 waves x 16 q-rows; k-tiles of 32.
__global__ __launch_bounds__(256) void k_attn(const u16* __restrict__ Q,
                                              const u16* __restrict__ Kg,
                                              const u16* __restrict__ Vg,
                                              u16* __restrict__ O) {
  __shared__ __align__(16) u16 Qt[128][136];
  __shared__ __align__(16) u16 Kt[32][136];
  __shared__ __align__(16) u16 Vt[64][40];
  __shared__ __align__(16) u16 Ps[8][16][40];

  const int tid = threadIdx.x;
  const int lane = tid & 31, wid = tid >> 5;
  const int l15 = lane & 15, half = lane >> 4;
  const int q0 = blockIdx.x * 128;
  const int h = blockIdx.y;
  const int b = blockIdx.z;
  const size_t baseQK = ((size_t)b * TT) * (HH * 128) + h * 128;
  const size_t baseV  = ((size_t)b * TT) * DD + h * DH;

  { // stage Q tile: pure b128 copies
    int c = tid & 15, q = tid >> 4;
    #pragma unroll
    for (int i = 0; i < 8; ++i) {
      int qq = q + 16 * i;
      *(uint4*)&Qt[qq][c * 8] =
          *(const uint4*)(Q + baseQK + (size_t)(q0 + qq) * (HH * 128) + c * 8);
    }
  }
  __syncthreads();

  const int wq = wid * 16;
  v16bf qf[4];
  #pragma unroll
  for (int dc = 0; dc < 4; ++dc)
    qf[dc] = load_frag_a(&Qt[wq][dc * 32], 136, l15, half);

  float m_j[8], l_j[8];
  v8f oacc[4];
  #pragma unroll
  for (int j = 0; j < 8; ++j) { m_j[j] = -1e30f; l_j[j] = 0.f; }
  #pragma unroll
  for (int vt = 0; vt < 4; ++vt)
    #pragma unroll
    for (int j = 0; j < 8; ++j) oacc[vt][j] = 0.f;

  for (int kt = 0; kt < q0 + 128; kt += 32) {
    __syncthreads();
    { // K tile copy [32 keys][128 dims]
      int c = tid & 15, kk = tid >> 4;
      #pragma unroll
      for (int i = 0; i < 2; ++i) {
        int r = kk + 16 * i;
        *(uint4*)&Kt[r][c * 8] =
            *(const uint4*)(Kg + baseQK + (size_t)(kt + r) * (HH * 128) + c * 8);
      }
    }
    { // V tile transpose [64 dv][32 keys] (small, element-wise)
      int dv = tid & 63, kr = tid >> 6;
      #pragma unroll
      for (int i = 0; i < 8; ++i) {
        int kk = kr + 4 * i;
        Vt[dv][kk] = Vg[baseV + (size_t)(kt + kk) * DD + dv];
      }
    }
    __syncthreads();

    if (kt <= q0 + wq + 15) {
      v8f s[2];
      #pragma unroll
      for (int nt = 0; nt < 2; ++nt)
        #pragma unroll
        for (int j = 0; j < 8; ++j) s[nt][j] = 0.f;
      #pragma unroll
      for (int nt = 0; nt < 2; ++nt)
        #pragma unroll
        for (int dc = 0; dc < 4; ++dc) {
          v16bf kf = load_frag_b(&Kt[nt * 16][dc * 32], 136, l15, half);
          s[nt] = wmma_bf16(qf[dc], kf, s[nt]);
        }

      const bool needmask = (kt + 31 > q0 + wq);
      #pragma unroll
      for (int j = 0; j < 8; ++j) {
        const int row = q0 + wq + j + half * 8;
        bool msk[2]; float sv[2];
        #pragma unroll
        for (int nt = 0; nt < 2; ++nt) {
          int col = kt + nt * 16 + l15;
          msk[nt] = needmask && (col > row);
          sv[nt] = msk[nt] ? -1e30f : s[nt][j];
        }
        float rmax = fmaxf(sv[0], sv[1]);
        for (int m = 8; m > 0; m >>= 1) rmax = fmaxf(rmax, __shfl_xor(rmax, m, 32));
        float mn = fmaxf(m_j[j], rmax);
        float alpha = expf(m_j[j] - mn);
        m_j[j] = mn;
        float psum = 0.f;
        #pragma unroll
        for (int nt = 0; nt < 2; ++nt) {
          float p = msk[nt] ? 0.f : expf(sv[nt] - mn);
          s[nt][j] = p;
          psum += p;
        }
        for (int m = 8; m > 0; m >>= 1) psum += __shfl_xor(psum, m, 32);
        l_j[j] = l_j[j] * alpha + psum;
        #pragma unroll
        for (int vt = 0; vt < 4; ++vt) oacc[vt][j] *= alpha;
      }

      // P -> per-wave LDS region, reload in A-operand layout
      #pragma unroll
      for (int j = 0; j < 8; ++j)
        #pragma unroll
        for (int nt = 0; nt < 2; ++nt)
          Ps[wid][j + half * 8][nt * 16 + l15] = f2bf(s[nt][j]);
      asm volatile("s_wait_dscnt 0x0" ::: "memory");
      v16bf pf = load_frag_a(&Ps[wid][0][0], 40, l15, half);
      #pragma unroll
      for (int vt = 0; vt < 4; ++vt) {
        v16bf vf = load_frag_b(&Vt[vt * 16][0], 40, l15, half);
        oacc[vt] = wmma_bf16(pf, vf, oacc[vt]);
      }
    }
  }

  #pragma unroll
  for (int vt = 0; vt < 4; ++vt)
    #pragma unroll
    for (int j = 0; j < 8; ++j) {
      int q = q0 + wq + j + half * 8;
      int dv = vt * 16 + l15;
      O[baseV + (size_t)q * DD + dv] = f2bf(oacc[vt][j] / l_j[j]);
    }
}

// ---------------- launcher ----------------
extern "C" void kernel_launch(void* const* d_in, const int* in_sizes, int n_in,
                              void* d_out, int out_size, void* d_ws, size_t ws_size,
                              hipStream_t stream) {
  (void)n_in; (void)out_size; (void)ws_size;

  // Detect flattening: insertion order (idx4 = lie_W, 32768) vs sorted pytree order.
  bool sorted = (in_sizes[4] != 32768);

  int I_aemb, I_oemb, I_lieW, I_lieb, I_posW, I_posb;
  int I_Wqc, I_bqc, I_Wkc, I_bkc, I_Wqs, I_bqs, I_Wks, I_bks, I_Wv, I_bv, I_Wo, I_bo;
  int I_l1g, I_l1b, I_l2g, I_l2b, I_W1, I_b1, I_W2, I_b2;
  int I_og, I_ob, I_oW, I_obias;
  if (!sorted) {
    I_aemb=2; I_oemb=3; I_lieW=4; I_lieb=5; I_posW=6; I_posb=7;
    I_Wqc=8; I_bqc=9; I_Wkc=10; I_bkc=11; I_Wqs=12; I_bqs=13; I_Wks=14; I_bks=15;
    I_Wv=16; I_bv=17; I_Wo=18; I_bo=19; I_l1g=20; I_l1b=21; I_l2g=22; I_l2b=23;
    I_W1=24; I_b1=25; I_W2=26; I_b2=27; I_og=28; I_ob=29; I_oW=30; I_obias=31;
  } else {
    I_aemb=2; I_W1=3; I_W2=4; I_Wkc=5; I_Wks=6; I_Wo=7; I_Wqc=8; I_Wqs=9; I_Wv=10;
    I_b1=11; I_b2=12; I_bkc=13; I_bks=14; I_bo=15; I_bqc=16; I_bqs=17; I_bv=18;
    I_l1b=19; I_l1g=20; I_l2b=21; I_l2g=22;
    I_lieW=23; I_lieb=24; I_oemb=25; I_oW=26; I_obias=27; I_ob=28; I_og=29;
    I_posW=30; I_posb=31;
  }

  const int* actions = (const int*)d_in[0];
  const int* obs     = (const int*)d_in[1];
  const float* aemb  = (const float*)d_in[I_aemb];
  const float* oemb  = (const float*)d_in[I_oemb];
  const float* lieW  = (const float*)d_in[I_lieW];
  const float* lieb  = (const float*)d_in[I_lieb];
  const float* posW  = (const float*)d_in[I_posW];
  const float* posb  = (const float*)d_in[I_posb];
  const float* Wqc = (const float*)d_in[I_Wqc]; const float* bqc = (const float*)d_in[I_bqc];
  const float* Wkc = (const float*)d_in[I_Wkc]; const float* bkc = (const float*)d_in[I_bkc];
  const float* Wqs = (const float*)d_in[I_Wqs]; const float* bqs = (const float*)d_in[I_bqs];
  const float* Wks = (const float*)d_in[I_Wks]; const float* bks = (const float*)d_in[I_bks];
  const float* Wv  = (const float*)d_in[I_Wv];  const float* bv  = (const float*)d_in[I_bv];
  const float* Wo  = (const float*)d_in[I_Wo];  const float* bo  = (const float*)d_in[I_bo];
  const float* l1g = (const float*)d_in[I_l1g]; const float* l1b = (const float*)d_in[I_l1b];
  const float* l2g = (const float*)d_in[I_l2g]; const float* l2b = (const float*)d_in[I_l2b];
  const float* W1 = (const float*)d_in[I_W1];   const float* b1 = (const float*)d_in[I_b1];
  const float* W2 = (const float*)d_in[I_W2];   const float* b2 = (const float*)d_in[I_b2];
  const float* outg = (const float*)d_in[I_og]; const float* outb = (const float*)d_in[I_ob];
  const float* outW = (const float*)d_in[I_oW]; const float* outbias = (const float*)d_in[I_obias];
  float* out = (float*)d_out;

  // ---------------- workspace layout ----------------
  float* x     = (float*)d_ws;                        // 4096x1024 f32 residual
  float* delta = x + (size_t)MM * DD;                 // 4096x32 f32
  u16* u = (u16*)(delta + (size_t)MM * RR);
  auto take = [&](size_t n) { u16* p = u; u += n; return p; };
  u16* hbuf  = take((size_t)MM * DD);
  u16* posb16= take((size_t)MM * DD);
  u16* Qb    = take((size_t)MM * 2048);
  u16* Kb    = take((size_t)MM * 2048);
  u16* mid   = Qb;                                    // 4096x4096 reuses Q|K region
  u16* Vb    = take((size_t)MM * DD);
  u16* ao    = take((size_t)MM * DD);
  u16* cs    = take((size_t)MM * 64);
  u16* wefft = take((size_t)DD * 64);
  // pre-transposed bf16 weights
  u16 *wt_qc[2], *wt_kc[2], *wt_qs[2], *wt_ks[2], *wt_v[2], *wt_o[2], *wt_1[2], *wt_2[2];
  for (int i = 0; i < 2; ++i) {
    wt_qc[i] = take((size_t)DD * DD); wt_kc[i] = take((size_t)DD * DD);
    wt_qs[i] = take((size_t)DD * DD); wt_ks[i] = take((size_t)DD * DD);
    wt_v[i]  = take((size_t)DD * DD); wt_o[i]  = take((size_t)DD * DD);
    wt_1[i]  = take((size_t)DD * 4 * DD); wt_2[i] = take((size_t)4 * DD * DD);
  }
  u16* wt_out = take((size_t)DD * OV);

  auto wtrans = [&](const float* W, u16* Wt, int N, int K) {
    k_wtrans<<<dim3(N / 32, K / 32), 256, 0, stream>>>(W, Wt, N, K);
  };
  for (int i = 0; i < 2; ++i) {
    wtrans(Wqc + (size_t)i * DD * DD, wt_qc[i], DD, DD);
    wtrans(Wkc + (size_t)i * DD * DD, wt_kc[i], DD, DD);
    wtrans(Wqs + (size_t)i * DD * DD, wt_qs[i], DD, DD);
    wtrans(Wks + (size_t)i * DD * DD, wt_ks[i], DD, DD);
    wtrans(Wv  + (size_t)i * DD * DD, wt_v[i],  DD, DD);
    wtrans(Wo  + (size_t)i * DD * DD, wt_o[i],  DD, DD);
    wtrans(W1 + (size_t)i * DD * 4 * DD, wt_1[i], 4 * DD, DD);
    wtrans(W2 + (size_t)i * 4 * DD * DD, wt_2[i], DD, 4 * DD);
  }
  wtrans(outW, wt_out, OV, DD);

  auto gemm = [&](const u16* A, int lda, const u16* Wt, const float* bias,
                  const float* res, void* C, int ldc, int N, int K,
                  int mode, int hsplit, int qoff, float scale, int obf) {
    dim3 g(N / 128, MM / 128);
    k_gemm<<<g, 256, 0, stream>>>(A, lda, Wt, K, bias, res, C, ldc, K,
                                  mode, hsplit, qoff, scale, obf);
  };

  // position path
  k_embed<<<MM, 256, 0, stream>>>(obs, oemb, x);
  k_delta<<<MM, 256, 0, stream>>>(actions, aemb, lieW, lieb, delta);
  k_weff<<<64, 256, 0, stream>>>(posW, wefft);
  k_scan<<<1, 64, 0, stream>>>(delta, cs);
  gemm(cs, 64, wefft, posb, nullptr, posb16, DD, DD, 64, 0, 0, 0, 1.f, 1);

  for (int i = 0; i < 2; ++i) {
    k_ln<<<MM, 256, 0, stream>>>(x, l1g + i * DD, l1b + i * DD, hbuf);
    // Q = [Qc|Qs] per head (128 dims) with 1/sqrt(DH) folded in; K likewise (unscaled)
    gemm(hbuf,   DD, wt_qc[i], bqc + i * DD, nullptr, Qb, 2048, DD, DD, 0, 1, 0,  0.125f, 1);
    gemm(posb16, DD, wt_qs[i], bqs + i * DD, nullptr, Qb, 2048, DD, DD, 0, 1, 64, 0.125f, 1);
    gemm(hbuf,   DD, wt_kc[i], bkc + i * DD, nullptr, Kb, 2048, DD, DD, 0, 1, 0,  1.f, 1);
    gemm(posb16, DD, wt_ks[i], bks + i * DD, nullptr, Kb, 2048, DD, DD, 0, 1, 64, 1.f, 1);
    gemm(hbuf,   DD, wt_v[i],  bv + i * DD,  nullptr, Vb, DD,   DD, DD, 0, 0, 0,  1.f, 1);

    k_attn<<<dim3(TT / 128, HH, BB), 256, 0, stream>>>(Qb, Kb, Vb, ao);

    gemm(ao, DD, wt_o[i], bo + i * DD, x, x, DD, DD, DD, 2, 0, 0, 1.f, 0);     // x += O-proj
    k_ln<<<MM, 256, 0, stream>>>(x, l2g + i * DD, l2b + i * DD, hbuf);
    gemm(hbuf, DD, wt_1[i], b1 + i * 4 * DD, nullptr, mid, 4 * DD, 4 * DD, DD, 1, 0, 0, 1.f, 1);
    gemm(mid, 4 * DD, wt_2[i], b2 + i * DD, x, x, DD, DD, 4 * DD, 2, 0, 0, 1.f, 0); // x += ffn
  }

  k_ln<<<MM, 256, 0, stream>>>(x, outg, outb, hbuf);
  gemm(hbuf, DD, wt_out, outbias, nullptr, out, OV, OV, DD, 0, 0, 0, 1.f, 0);
}
